// EncoderBiRNN_4947802325447
// MI455X (gfx1250) — compile-verified, matrix operand order checked by reading, not run on hardware
//
#include <hip/hip_runtime.h>

// ---------------------------------------------------------------------------
// Types
// ---------------------------------------------------------------------------
typedef __attribute__((ext_vector_type(16))) __bf16 v16bf;
typedef __attribute__((ext_vector_type(8)))  float  v8f;
typedef unsigned short bfraw;   // raw bf16 bits in memory

union FragBF { v16bf v; uint4 q[2]; };

// Problem constants
#define VV   32000
#define EE   1024
#define HH2  1024
#define GG   3072      // 3*H2
#define BB   128
#define LL   512
#define MROWS (LL*BB)  // 65536

__device__ __forceinline__ unsigned short f2bf(float f) {
    unsigned u = __float_as_uint(f);
    unsigned r = (u + 0x7FFFu + ((u >> 16) & 1u)) >> 16;   // RNE
    return (unsigned short)r;
}
__device__ __forceinline__ float bf2f(bfraw v) {
    return __uint_as_float(((unsigned)v) << 16);
}

__device__ __forceinline__ v8f wmma_bf16(v16bf a, v16bf b, v8f c) {
    // D = A(16x32 bf16) * B(32x16 bf16) + C(16x16 f32)
    return __builtin_amdgcn_wmma_f32_16x16x32_bf16(
        false, a, false, b, (short)0, c, false, false);
}

// Load one 16x32 bf16 fragment (A or B) from a row-major matrix.
// ISA layout: lane&15 = row, lane>>4 selects K-half; two 16B chunks per lane.
__device__ __forceinline__ v16bf loadFrag(const bfraw* base, int row0, int ld,
                                          int k, int lane) {
    const bfraw* p = base + (size_t)(row0 + (lane & 15)) * ld
                          + k + ((lane >> 4) << 3);
    FragBF f;
    f.q[0] = *(const uint4*)(p);        // K-chunk 0 (VGPRs 0-3)
    f.q[1] = *(const uint4*)(p + 16);   // K-chunk 1 (VGPRs 4-7)
    return f.v;
}

// gfx1250 async copy: 16B per lane, global -> LDS, tracked by ASYNCcnt.
// VDST = per-lane LDS byte address, VADDR = per-lane 64-bit global address.
__device__ __forceinline__ void async_copy_b128(unsigned lds_addr, const void* g) {
    asm volatile("global_load_async_to_lds_b128 %0, %1, off"
                 :: "v"(lds_addr), "v"(g) : "memory");
}
__device__ __forceinline__ void wait_asynccnt0() {
    asm volatile("s_wait_asynccnt 0x0" ::: "memory");
}

// ---------------------------------------------------------------------------
// Kernel 1: embedding gather + fp32 -> bf16, x laid out (L, B, E)
// ---------------------------------------------------------------------------
__global__ __launch_bounds__(256)
void gather_embed_kernel(const int* __restrict__ seq,
                         const float* __restrict__ emb,
                         bfraw* __restrict__ xbf) {
    int row = blockIdx.x;               // row = l*B + b
    int l = row >> 7;
    int b = row & 127;
    int tok = seq[b * LL + l];          // seq is (B, L)
    const float* src = emb + (size_t)tok * EE;
    bfraw* dst = xbf + (size_t)row * EE;
    int e = threadIdx.x << 2;           // 256 threads * 4 = 1024
    float4 v = *(const float4*)(src + e);
    ushort4 o = make_ushort4(f2bf(v.x), f2bf(v.y), f2bf(v.z), f2bf(v.w));
    *(ushort4*)(dst + e) = o;
}

// ---------------------------------------------------------------------------
// Kernel 2: generic fp32 -> bf16 (weights)
// ---------------------------------------------------------------------------
__global__ __launch_bounds__(256)
void cvt_bf16_kernel(const float* __restrict__ src, bfraw* __restrict__ dst, int n) {
    int i = (blockIdx.x * 256 + threadIdx.x) << 2;
    if (i + 3 < n) {
        float4 v = *(const float4*)(src + i);
        ushort4 o = make_ushort4(f2bf(v.x), f2bf(v.y), f2bf(v.z), f2bf(v.w));
        *(ushort4*)(dst + i) = o;
    }
}

// ---------------------------------------------------------------------------
// Kernel 3: gx = X(65536x1024) * W^T(1024x3072) + bias, output bf16.
//   WG = 256 thr (8 waves) -> 128x256 block; wave = 64x64 tile (4x4 WMMA).
//   K sliced by 32; A(128x32) and B(256x32) staged in LDS via
//   global_load_async_to_lds_b128 (double buffered, s_wait_asynccnt).
//   LDS rows padded to 40 elems (20-dword stride -> conflict-free b128).
// ---------------------------------------------------------------------------
#define T_LD 40          // padded LDS row stride (elements)

__global__ __launch_bounds__(256)
void gx_gemm_kernel(const bfraw* __restrict__ X,
                    const bfraw* __restrict__ W,
                    const float* __restrict__ bias,
                    bfraw* __restrict__ out) {
    __shared__ bfraw sA[2][128 * T_LD];   // 2 x 10240 B
    __shared__ bfraw sB[2][256 * T_LD];   // 2 x 20480 B

    const int lane = threadIdx.x & 31;
    const int wv   = threadIdx.x >> 5;        // 0..7
    const int mb   = blockIdx.x / (GG / 256); // 0..511
    const int nb   = blockIdx.x % (GG / 256); // 0..11
    const int m0   = mb * 128;
    const int n0   = nb * 256;
    const int m0w  = (wv >> 2) * 64;          // wave tile inside block
    const int n0w  = (wv & 3) * 64;

    // Issue async copies of the K-slice [kk, kk+32) into buffer `buf`.
    // 48 wave-chunks of 512B: 16 for A (128x64B), 32 for B (256x64B).
    auto copy_tiles = [&](int kk, int buf) {
        #pragma unroll
        for (int i = 0; i < 6; ++i) {
            int c = wv * 6 + i;                 // uniform per wave
            if (c < 16) {
                int u = c * 32 + lane;          // 0..511
                int row = u >> 2, col = (u & 3) << 3;
                const bfraw* g = X + (size_t)(m0 + row) * EE + kk + col;
                unsigned l = (unsigned)(size_t)(&sA[buf][row * T_LD + col]);
                async_copy_b128(l, g);
            } else {
                int u = (c - 16) * 32 + lane;   // 0..1023
                int row = u >> 2, col = (u & 3) << 3;
                const bfraw* g = W + (size_t)(n0 + row) * EE + kk + col;
                unsigned l = (unsigned)(size_t)(&sB[buf][row * T_LD + col]);
                async_copy_b128(l, g);
            }
        }
    };

    v8f acc[4][4];
    #pragma unroll
    for (int i = 0; i < 4; ++i)
        #pragma unroll
        for (int j = 0; j < 4; ++j)
            acc[i][j] = v8f{0.f,0.f,0.f,0.f,0.f,0.f,0.f,0.f};

    copy_tiles(0, 0);
    wait_asynccnt0();
    __syncthreads();

    for (int kk = 0; kk < EE; kk += 32) {
        int buf = (kk >> 5) & 1;
        if (kk + 32 < EE) copy_tiles(kk + 32, buf ^ 1);

        v16bf fa[4], fb[4];
        #pragma unroll
        for (int f = 0; f < 4; ++f)
            fa[f] = loadFrag(sA[buf], m0w + f * 16, T_LD, 0, lane);
        #pragma unroll
        for (int f = 0; f < 4; ++f)
            fb[f] = loadFrag(sB[buf], n0w + f * 16, T_LD, 0, lane);
        #pragma unroll
        for (int i = 0; i < 4; ++i)
            #pragma unroll
            for (int j = 0; j < 4; ++j)
                acc[i][j] = wmma_bf16(fa[i], fb[j], acc[i][j]);

        wait_asynccnt0();
        __syncthreads();
    }

    // Epilogue: C layout lane&15 = N, VGPR v = M row (lanes 16-31: M = v+8)
    #pragma unroll
    for (int fm = 0; fm < 4; ++fm) {
        #pragma unroll
        for (int fn = 0; fn < 4; ++fn) {
            int n = n0 + n0w + fn * 16 + (lane & 15);
            int m = m0 + m0w + fm * 16 + ((lane >> 4) << 3);
            float bv = bias[n];
            bfraw* o = out + (size_t)m * GG + n;
            #pragma unroll
            for (int v = 0; v < 8; ++v)
                o[(size_t)v * GG] = f2bf(acc[fm][fn][v] + bv);
        }
    }
}

// ---------------------------------------------------------------------------
// Kernel 4: persistent bidirectional GRU scan (unchanged structure).
//   grid = (32 j-blocks, 2 directions), block = 384 threads (12 waves).
//   Whh slice (96x1024 bf16 ~192KB) LDS-resident; per step 128x96 GEMM
//   (wave = 32x32 tile, B-frags via ds_load_b128), fused gates, h fp32 +
//   bf16 ping-pong, per-direction grid barrier between steps.
// ---------------------------------------------------------------------------
#define W_LD  1032      // 1024 + 8 bf16 pad (keeps 16B alignment)
#define GH_LD 100       // 96 + 4 f32 pad
#define SCAN_SMEM (96 * W_LD * 2 + 128 * GH_LD * 4)

__global__ __launch_bounds__(384)
void gru_scan_kernel(const bfraw* __restrict__ whh_f, const bfraw* __restrict__ whh_r,
                     const float* __restrict__ bhh_f, const float* __restrict__ bhh_r,
                     const bfraw* __restrict__ gx_f,  const bfraw* __restrict__ gx_r,
                     float* __restrict__ h32, bfraw* __restrict__ hb,
                     unsigned* __restrict__ ctr, float* __restrict__ out) {
    extern __shared__ unsigned char smem[];
    bfraw* ldsW  = (bfraw*)smem;                               // [96][W_LD]
    float* ldsGH = (float*)(smem + 96 * W_LD * sizeof(bfraw)); // [128][GH_LD]

    const int dir = blockIdx.y;
    const int j0  = blockIdx.x * 32;
    const int tid = threadIdx.x;
    const int lane = tid & 31;
    const int wv   = tid >> 5;            // 0..11

    const bfraw* whh = dir ? whh_r : whh_f;
    const float* bhh = dir ? bhh_r : bhh_f;
    const bfraw* gx  = dir ? gx_r  : gx_f;
    float* h32d = h32 + (size_t)dir * BB * HH2;

    // Preload Whh slice: gate rows {j, H2+j, 2*H2+j} for j in [j0, j0+32)
    for (int c = tid; c < 96 * 128; c += 384) {
        int rr  = c >> 7;
        int off = (c & 127) << 3;
        int gate = rr >> 5, jj = rr & 31;
        const uint4* src = (const uint4*)(whh + ((size_t)(gate * HH2 + j0 + jj)) * HH2 + off);
        *(uint4*)(ldsW + rr * W_LD + off) = *src;
    }
    __syncthreads();

    const int m0 = (wv / 3) * 32;
    const int n0 = (wv % 3) * 32;

    for (int t = 0; t < LL; ++t) {
        int cur = t & 1, nxt = cur ^ 1;
        const bfraw* hin  = hb + (size_t)(cur * 2 + dir) * BB * HH2;
        bfraw*       hout = hb + (size_t)(nxt * 2 + dir) * BB * HH2;

        v8f a00 = {0.f,0.f,0.f,0.f,0.f,0.f,0.f,0.f};
        v8f a01 = a00, a10 = a00, a11 = a00;
        for (int k = 0; k < HH2; k += 32) {
            v16bf fa0 = loadFrag(hin,  m0,      HH2,  k, lane);
            v16bf fa1 = loadFrag(hin,  m0 + 16, HH2,  k, lane);
            v16bf fb0 = loadFrag(ldsW, n0,      W_LD, k, lane);
            v16bf fb1 = loadFrag(ldsW, n0 + 16, W_LD, k, lane);
            a00 = wmma_bf16(fa0, fb0, a00);
            a01 = wmma_bf16(fa0, fb1, a01);
            a10 = wmma_bf16(fa1, fb0, a10);
            a11 = wmma_bf16(fa1, fb1, a11);
        }
        #pragma unroll
        for (int fm = 0; fm < 2; ++fm) {
            #pragma unroll
            for (int fn = 0; fn < 2; ++fn) {
                v8f acc = fm == 0 ? (fn == 0 ? a00 : a01) : (fn == 0 ? a10 : a11);
                int n = n0 + fn * 16 + (lane & 15);
                int m = m0 + fm * 16 + ((lane >> 4) << 3);
                float* p = ldsGH + m * GH_LD + n;
                #pragma unroll
                for (int v = 0; v < 8; ++v) p[v * GH_LD] = acc[v];
            }
        }
        __syncthreads();

        // Fused gate math for this WG's 32 hidden columns
        int t_eff = dir ? (LL - 1 - t) : t;
        for (int idx = tid; idx < BB * 32; idx += 384) {
            int b = idx >> 5, jj = idx & 31;
            int j = j0 + jj;
            float ghr = ldsGH[b * GH_LD + jj]       + bhh[j];
            float ghz = ldsGH[b * GH_LD + 32 + jj]  + bhh[HH2 + j];
            float ghn = ldsGH[b * GH_LD + 64 + jj]  + bhh[2 * HH2 + j];
            size_t gxo = ((size_t)t_eff * BB + b) * GG;
            float r  = 1.f / (1.f + __expf(-(bf2f(gx[gxo + j])       + ghr)));
            float z  = 1.f / (1.f + __expf(-(bf2f(gx[gxo + HH2 + j]) + ghz)));
            float nn = tanhf(bf2f(gx[gxo + 2 * HH2 + j]) + r * ghn);
            float ho = h32d[b * HH2 + j];
            float hn = (1.f - z) * nn + z * ho;
            h32d[b * HH2 + j] = hn;
            hout[b * HH2 + j] = f2bf(hn);
            if (t == LL - 1)
                out[(size_t)b * (2 * HH2) + dir * HH2 + j] = hn;
        }
        __syncthreads();

        // Per-direction grid barrier (32 WGs)
        if (t < LL - 1) {
            if (tid == 0) {
                __builtin_amdgcn_fence(__ATOMIC_RELEASE, "agent");
                __hip_atomic_fetch_add(&ctr[dir], 1u, __ATOMIC_RELAXED,
                                       __HIP_MEMORY_SCOPE_AGENT);
                unsigned target = 32u * (unsigned)(t + 1);
                while (__hip_atomic_load(&ctr[dir], __ATOMIC_RELAXED,
                                         __HIP_MEMORY_SCOPE_AGENT) < target)
                    __builtin_amdgcn_s_sleep(1);
                __builtin_amdgcn_fence(__ATOMIC_ACQUIRE, "agent");
            }
            __syncthreads();
        }
    }
}

// ---------------------------------------------------------------------------
// Launch
// ---------------------------------------------------------------------------
extern "C" void kernel_launch(void* const* d_in, const int* in_sizes, int n_in,
                              void* d_out, int out_size, void* d_ws, size_t ws_size,
                              hipStream_t stream) {
    const int*   seq   = (const int*)  d_in[0];
    const float* emb   = (const float*)d_in[1];
    const float* Wih_f = (const float*)d_in[2];
    const float* Whh_f = (const float*)d_in[3];
    const float* bih_f = (const float*)d_in[4];
    const float* bhh_f = (const float*)d_in[5];
    const float* Wih_r = (const float*)d_in[6];
    const float* Whh_r = (const float*)d_in[7];
    const float* bih_r = (const float*)d_in[8];
    const float* bhh_r = (const float*)d_in[9];
    float* out = (float*)d_out;

    char* ws = (char*)d_ws;
    size_t off = 0;
    bfraw* xbf     = (bfraw*)(ws + off); off += (size_t)MROWS * EE * 2;   // 134 MB
    bfraw* wihf_bf = (bfraw*)(ws + off); off += (size_t)GG * EE * 2;
    bfraw* whhf_bf = (bfraw*)(ws + off); off += (size_t)GG * HH2 * 2;
    bfraw* wihr_bf = (bfraw*)(ws + off); off += (size_t)GG * EE * 2;
    bfraw* whhr_bf = (bfraw*)(ws + off); off += (size_t)GG * HH2 * 2;
    bfraw* gxf     = (bfraw*)(ws + off); off += (size_t)MROWS * GG * 2;   // 403 MB
    bfraw* gxr     = (bfraw*)(ws + off); off += (size_t)MROWS * GG * 2;   // 403 MB
    float* h32     = (float*)(ws + off); off += (size_t)2 * BB * HH2 * 4;
    bfraw* hb      = (bfraw*)(ws + off); off += (size_t)4 * BB * HH2 * 2; // [buf][dir]
    unsigned* ctr  = (unsigned*)(ws + off); off += 256;

    // 1) embedding gather -> x bf16, layout (L, B, E)
    gather_embed_kernel<<<MROWS, 256, 0, stream>>>(seq, emb, xbf);

    // 2) weights -> bf16
    const int WN = GG * EE;                       // 3,145,728
    cvt_bf16_kernel<<<WN / 1024, 256, 0, stream>>>(Wih_f, wihf_bf, WN);
    cvt_bf16_kernel<<<WN / 1024, 256, 0, stream>>>(Whh_f, whhf_bf, WN);
    cvt_bf16_kernel<<<WN / 1024, 256, 0, stream>>>(Wih_r, wihr_bf, WN);
    cvt_bf16_kernel<<<WN / 1024, 256, 0, stream>>>(Wih_r == Wih_r ? Whh_r : Whh_r, whhr_bf, WN);

    // 3) input-projection GEMMs: 512 m-blocks x 12 n-blocks = 6144 WGs
    const int gemm_blocks = (MROWS / 128) * (GG / 256);
    gx_gemm_kernel<<<gemm_blocks, 256, 0, stream>>>(xbf, wihf_bf, bih_f, gxf);
    gx_gemm_kernel<<<gemm_blocks, 256, 0, stream>>>(xbf, wihr_bf, bih_r, gxr);

    // 4) zero recurrent state + barrier counters (graph-capturable memsets)
    hipMemsetAsync(h32, 0, (size_t)2 * BB * HH2 * 4, stream);
    hipMemsetAsync(hb,  0, (size_t)4 * BB * HH2 * 2, stream);
    hipMemsetAsync(ctr, 0, 256, stream);

    // 5) persistent bidirectional scan
    hipFuncSetAttribute((const void*)gru_scan_kernel,
                        hipFuncAttributeMaxDynamicSharedMemorySize, SCAN_SMEM);
    gru_scan_kernel<<<dim3(32, 2), 384, SCAN_SMEM, stream>>>(
        whhf_bf, whhr_bf, bhh_f, bhh_r, gxf, gxr, h32, hb, ctr, out);
}
